// MultiHeadAttention_67448166416726
// MI455X (gfx1250) — compile-verified, matrix operand order checked by reading, not run on hardware
//
#include <hip/hip_runtime.h>
#include <hip/hip_bf16.h>
#include <hip/hip_fp16.h>

typedef __attribute__((ext_vector_type(16))) _Float16 v16h;
typedef __attribute__((ext_vector_type(8)))  _Float16 v8h;
typedef __attribute__((ext_vector_type(8)))  float    v8f;

#define B_   8
#define C_   64
#define N_   1024
#define F_   64
#define HEADS_ 8
#define DH_  8
#define SCALE_ 0.35355339059327373f  // 1/sqrt(8)

// Build one A-matrix fragment (16-bit A layout, 16x32 tile) from an f32
// row-major matrix. Per ISA: lane<16 holds K = kbase..kbase+7 then
// kbase+16..kbase+23; lane>=16 uses kbase+8 (caller passes adjusted kbase).
__device__ inline v16h load_wtile_f32(const float* __restrict__ W, int ldw,
                                      int row, int kbase) {
    v16h a;
    const float* p = W + (size_t)row * ldw;
#pragma unroll
    for (int j = 0; j < 8; ++j) a[j]     = (_Float16)p[kbase + j];
#pragma unroll
    for (int j = 0; j < 8; ++j) a[8 + j] = (_Float16)p[kbase + 16 + j];
    return a;
}

// -------- Kernel 1: QKV projections (3x GEMM 64x64x1024 per batch) --------
// Writes Qt/Kt as [B][N][F] f16 (scaled q), V as [B][F][N] f16.
__global__ __launch_bounds__(256) void qkv_proj_kernel(
    const float* __restrict__ x,
    const float* __restrict__ Wq, const float* __restrict__ bq,
    const float* __restrict__ Wk, const float* __restrict__ bk,
    const float* __restrict__ Wv, const float* __restrict__ bv,
    _Float16* __restrict__ Qt, _Float16* __restrict__ Kt,
    _Float16* __restrict__ V)
{
    int lane = threadIdx.x & 31;
    int wv   = threadIdx.x >> 5;
    int w    = blockIdx.x * 8 + wv;          // 6144 wave-tasks
    int nt   = w & 63;  w >>= 6;             // n tile 0..63
    int ft   = w & 3;   w >>= 2;             // f tile 0..3
    int mat  = w % 3;                        // 0=q 1=k 2=v
    int b    = w / 3;
    const float* W    = (mat == 0) ? Wq : ((mat == 1) ? Wk : Wv);
    const float* bias = (mat == 0) ? bq : ((mat == 1) ? bk : bv);
    int f0 = ft * 16, n0 = nt * 16;
    int l16 = lane & 15, half16 = lane >> 4;

    v8f acc = {};
#pragma unroll
    for (int c0 = 0; c0 < C_; c0 += 32) {
        int kbase = c0 + (half16 ? 8 : 0);
        v16h a = load_wtile_f32(W, C_, f0 + l16, kbase);
        // B fragment: x[b][c][n], lane column n = n0+l16 (coalesced over lanes)
        v16h bm;
        const float* xb = x + (size_t)b * C_ * N_ + (n0 + l16);
#pragma unroll
        for (int j = 0; j < 8; ++j) bm[j]     = (_Float16)xb[(size_t)(kbase + j) * N_];
#pragma unroll
        for (int j = 0; j < 8; ++j) bm[8 + j] = (_Float16)xb[(size_t)(kbase + 16 + j) * N_];
        acc = __builtin_amdgcn_wmma_f32_16x16x32_f16(false, a, false, bm,
                                                     (short)0, acc, false, false);
    }
    int n = n0 + l16;
    if (mat < 2) {
        v8h hv;
#pragma unroll
        for (int r = 0; r < 8; ++r) {
            int f = f0 + r + 8 * half16;
            float val = acc[r] + bias[f];
            if (mat == 0) val *= SCALE_;
            hv[r] = (_Float16)val;
        }
        _Float16* dst = ((mat == 0) ? Qt : Kt) +
                        ((size_t)b * N_ + n) * F_ + f0 + 8 * half16;
        *(v8h*)dst = hv;                      // rows r contiguous in f
    } else {
#pragma unroll
        for (int r = 0; r < 8; ++r) {
            int f = f0 + r + 8 * half16;
            V[((size_t)b * F_ + f) * N_ + n] = (_Float16)(acc[r] + bias[f]);
        }
    }
}

// -------- Kernel 2: fused flash attention per (b, h, 16-query tile) --------
__global__ __launch_bounds__(256) void attention_kernel(
    const _Float16* __restrict__ Qt, const _Float16* __restrict__ Kt,
    const _Float16* __restrict__ V, _Float16* __restrict__ O)
{
    __shared__ __align__(16) _Float16 sc[8][16][32];  // 1KB per wave
    int lane = threadIdx.x & 31;
    int wv   = threadIdx.x >> 5;
    int w    = blockIdx.x * 8 + wv;          // 4096 wave-tasks
    int qt   = w & 63;  w >>= 6;
    int h    = w & 7;
    int b    = w >> 3;
    int n0 = qt * 16;
    int l16 = lane & 15, half16 = lane >> 4;

    // Q fragment: lanes>=16 all zero (K=8..15,24..31 are the d-padding)
    v16h aq = {};
    if (half16 == 0) {
        v8h q = *(const v8h*)(Qt + ((size_t)b * N_ + n0 + l16) * F_ + h * DH_);
#pragma unroll
        for (int j = 0; j < 8; ++j) aq[j] = q[j];
    }

    float rowmax[8], rowsum[8];
#pragma unroll
    for (int r = 0; r < 8; ++r) { rowmax[r] = -1e30f; rowsum[r] = 0.0f; }
    v8f acc = {};

    for (int m0 = 0; m0 < N_; m0 += 32) {
        // K fragments for the two 16-wide key subtiles
        v16h bk0 = {}, bk1 = {};
        if (half16 == 0) {
            v8h k0 = *(const v8h*)(Kt + ((size_t)b * N_ + m0 + l16) * F_ + h * DH_);
            v8h k1 = *(const v8h*)(Kt + ((size_t)b * N_ + m0 + 16 + l16) * F_ + h * DH_);
#pragma unroll
            for (int j = 0; j < 8; ++j) { bk0[j] = k0[j]; bk1[j] = k1[j]; }
        }
        v8f z = {};
        v8f S0 = __builtin_amdgcn_wmma_f32_16x16x32_f16(false, aq, false, bk0,
                                                        (short)0, z, false, false);
        v8f S1 = __builtin_amdgcn_wmma_f32_16x16x32_f16(false, aq, false, bk1,
                                                        (short)0, z, false, false);
        // online softmax; row r lives across the lane's 16-lane group
#pragma unroll
        for (int r = 0; r < 8; ++r) {
            float t = fmaxf(S0[r], S1[r]);
#pragma unroll
            for (int off = 1; off < 16; off <<= 1)
                t = fmaxf(t, __shfl_xor(t, off, 16));
            float nm   = fmaxf(rowmax[r], t);
            float corr = __expf(rowmax[r] - nm);
            float p0   = __expf(S0[r] - nm);
            float p1   = __expf(S1[r] - nm);
            float ps   = p0 + p1;
#pragma unroll
            for (int off = 1; off < 16; off <<= 1)
                ps += __shfl_xor(ps, off, 16);
            rowsum[r] = rowsum[r] * corr + ps;
            acc[r]   *= corr;
            rowmax[r] = nm;
            int row = r + 8 * half16;        // D-layout row for this lane
            sc[wv][row][l16]      = (_Float16)p0;
            sc[wv][row][l16 + 16] = (_Float16)p1;
        }
        // transpose scores D-layout -> A-layout through LDS (same wave, in-order DS)
        asm volatile("s_wait_dscnt 0x0" ::: "memory");
        int kb = half16 ? 8 : 0;
        v8h lo = *(const v8h*)&sc[wv][l16][kb];
        v8h hi = *(const v8h*)&sc[wv][l16][kb + 16];
        v16h aw;
#pragma unroll
        for (int j = 0; j < 8; ++j) { aw[j] = lo[j]; aw[8 + j] = hi[j]; }
        // V^T fragment: N dim = d (cols 8..15 zero pad), K dim = m block
        v16h bvf = {};
        int d = l16;
        if (d < DH_) {
            int ksel = half16 * 8;
            const _Float16* vp = V + ((size_t)b * F_ + h * DH_ + d) * N_ + m0 + ksel;
            v8h p  = *(const v8h*)vp;
            v8h q2 = *(const v8h*)(vp + 16);
#pragma unroll
            for (int j = 0; j < 8; ++j) { bvf[j] = p[j]; bvf[8 + j] = q2[j]; }
        }
        acc = __builtin_amdgcn_wmma_f32_16x16x32_f16(false, aw, false, bvf,
                                                     (short)0, acc, false, false);
    }
    // finalize: divide by rowsum, store O[b][h*8+d][n] (rows contiguous in n)
    int d = l16;
    if (d < DH_) {
        v8h ov;
#pragma unroll
        for (int r = 0; r < 8; ++r) ov[r] = (_Float16)(acc[r] / rowsum[r]);
        _Float16* dst = O + ((size_t)b * F_ + h * DH_ + d) * N_ + n0 + 8 * half16;
        *(v8h*)dst = ov;
    }
}

// -------- Kernel 3: output projection Y = Wo * O + bo (f32 out) --------
__global__ __launch_bounds__(256) void out_proj_kernel(
    const _Float16* __restrict__ O, const float* __restrict__ Wo,
    const float* __restrict__ bo, float* __restrict__ Y)
{
    int lane = threadIdx.x & 31;
    int wv   = threadIdx.x >> 5;
    int w    = blockIdx.x * 8 + wv;          // 2048 wave-tasks
    int nt   = w & 63;  w >>= 6;
    int ft   = w & 3;
    int b    = w >> 2;
    int f0 = ft * 16, n0 = nt * 16;
    int l16 = lane & 15, half16 = lane >> 4;

    v8f acc = {};
#pragma unroll
    for (int c0 = 0; c0 < F_; c0 += 32) {
        int kbase = c0 + (half16 ? 8 : 0);
        v16h a = load_wtile_f32(Wo, F_, f0 + l16, kbase);
        v16h bm;
        const _Float16* ob = O + (size_t)b * F_ * N_ + (n0 + l16);
#pragma unroll
        for (int j = 0; j < 8; ++j) bm[j]     = ob[(size_t)(kbase + j) * N_];
#pragma unroll
        for (int j = 0; j < 8; ++j) bm[8 + j] = ob[(size_t)(kbase + 16 + j) * N_];
        acc = __builtin_amdgcn_wmma_f32_16x16x32_f16(false, a, false, bm,
                                                     (short)0, acc, false, false);
    }
    int n = n0 + l16;
#pragma unroll
    for (int r = 0; r < 8; ++r) {
        int f = f0 + r + 8 * half16;
        Y[((size_t)b * F_ + f) * N_ + n] = acc[r] + bo[f];
    }
}

extern "C" void kernel_launch(void* const* d_in, const int* in_sizes, int n_in,
                              void* d_out, int out_size, void* d_ws, size_t ws_size,
                              hipStream_t stream)
{
    (void)in_sizes; (void)n_in; (void)out_size; (void)ws_size;
    const float* x  = (const float*)d_in[0];
    const float* Wq = (const float*)d_in[1];
    const float* bq = (const float*)d_in[2];
    const float* Wk = (const float*)d_in[3];
    const float* bk = (const float*)d_in[4];
    const float* Wv = (const float*)d_in[5];
    const float* bv = (const float*)d_in[6];
    const float* Wo = (const float*)d_in[7];
    const float* bo = (const float*)d_in[8];
    float* Y = (float*)d_out;

    size_t planeElems = (size_t)B_ * F_ * N_;   // 524288 halves = 1 MB each
    _Float16* Qt = (_Float16*)d_ws;
    _Float16* Kt = Qt + planeElems;
    _Float16* V  = Kt + planeElems;
    _Float16* O  = V  + planeElems;

    qkv_proj_kernel<<<768, 256, 0, stream>>>(x, Wq, bq, Wk, bk, Wv, bv, Qt, Kt, V);
    attention_kernel<<<512, 256, 0, stream>>>(Qt, Kt, V, O);
    out_proj_kernel<<<256, 256, 0, stream>>>(O, Wo, bo, Y);
}
